// SSD_3659312136224
// MI455X (gfx1250) — matrix operand order repaired and checked by reading, not compiled, not run on hardware
//
#include <hip/hip_runtime.h>

// ---- CDNA5 WMMA types --------------------------------------------------
typedef __attribute__((ext_vector_type(16))) __bf16        v16bf;
typedef __attribute__((ext_vector_type(8)))  float         v8f;
typedef __attribute__((ext_vector_type(4)))  unsigned int  v4u;

union Q2BF { v4u q[2]; v16bf bf; };

static __device__ __forceinline__ v4u v4zero() { v4u z = {0u, 0u, 0u, 0u}; return z; }

// dst = { hi16(f_odd), hi16(f_even) }  (one v_perm_b32)
static __device__ __forceinline__ unsigned pack_hi_pair(unsigned u_even, unsigned u_odd) {
  return __builtin_amdgcn_perm(u_odd, u_even, 0x07060302u);
}
// fp32 pair -> packed bf16 hi dword + packed bf16 lo (residual) dword
static __device__ __forceinline__ void split_pair(float a0, float a1,
                                                  unsigned& hi, unsigned& lo) {
  const unsigned u0 = __float_as_uint(a0), u1 = __float_as_uint(a1);
  const float r0 = a0 - __uint_as_float(u0 & 0xFFFF0000u);
  const float r1 = a1 - __uint_as_float(u1 & 0xFFFF0000u);
  hi = pack_hi_pair(u0, u1);
  lo = pack_hi_pair(__float_as_uint(r0), __float_as_uint(r1));
}

#define TOTAL_ANCHORS 8732
#define BATCH 32

// ---- weight pre-pack: fp32 [O,K] -> packed bf16-pair hi[] / lo[] dwords ----
__global__ __launch_bounds__(256) void pack_weights(const float* __restrict__ w,
                                                    unsigned* __restrict__ hi,
                                                    unsigned* __restrict__ lo,
                                                    int npairs) {
  const int i = blockIdx.x * 256 + threadIdx.x;
  if (i < npairs) {
    unsigned h, l;
    split_pair(w[2 * i], w[2 * i + 1], h, l);
    hi[i] = h; lo[i] = l;
  }
}

// One 3x3 same-pad conv head as implicit GEMM (see round-1 notes).
// Block 128 (4 waves); block tile M=128, N=32; wave tile 32x32 = 2x2 WMMA
// subtiles; fp32 as bf16 hi+lo split (3-term WMMA).
// PRE=true: A read from pre-packed hi/lo dword arrays (pure b128 loads).
template <bool PRE>
__global__ __launch_bounds__(128) void ssd_conv3x3_wmma(
    const float* __restrict__ X,        // [32, C, H, W]
    const float* __restrict__ Wt,       // [O, C*9] fp32 (fallback path)
    const unsigned* __restrict__ Whi,   // [O, C*9/2] packed bf16 hi pairs
    const unsigned* __restrict__ Wlo,   // [O, C*9/2] packed bf16 lo pairs
    const float* __restrict__ Bias,     // [O]
    float* __restrict__ Out,            // full d_out (float)
    int C, int H, int Wsp, int O, int A, int num_cols,
    int anchor_base, long long out_base)
{
  constexpr int MT = 2, NT = 2;
  constexpr int CSTR = 20;                      // LDS dwords per column (16B-aligned)
  const int HW   = H * Wsp;
  const int NHW  = BATCH * HW;
  const int Ktot = C * 9;
  const int Kp   = Ktot >> 1;                   // K-pairs per output channel

  const int tid    = threadIdx.x;
  const int lane   = tid & 31;
  const int wave   = tid >> 5;
  const int m_lane = lane & 15;
  const int kgrp   = lane >> 4;

  // column-major planes: [nn 0..31][kp 0..15], each lane's half-fragment is
  // 8 dwords in two 4-dword runs (kp 4*kgrp+0..3, 4*kgrp+8..11) -> 2x ds_load_b128
  __shared__ __align__(16) unsigned bhi_p[32 * CSTR];
  __shared__ __align__(16) unsigned blo_p[32 * CSTR];

  const int s_base = blockIdx.x * (16 * NT);
  const int o_wave = blockIdx.y * 128 + wave * (MT * 16);

  // ---- cooperative-load coordinates ----
  const int  nn_ld   = tid & 31;
  const int  kp_base = tid >> 5;                // 0..3
  const int  s_ld    = s_base + nn_ld;
  const bool sv      = s_ld < NHW;
  int h_ld = 0, w_ld = 0, pos_off = 0;
  const float* Xn = X;
  if (sv) {
    int n   = s_ld / HW;
    int r   = s_ld - n * HW;
    h_ld    = r / Wsp;
    w_ld    = r - h_ld * Wsp;
    pos_off = r;
    Xn      = X + (size_t)n * C * HW;
  }

  v8f acc[MT][NT];
  #pragma unroll
  for (int mt = 0; mt < MT; ++mt)
    #pragma unroll
    for (int nt = 0; nt < NT; ++nt)
      #pragma unroll
      for (int e = 0; e < 8; ++e) acc[mt][nt][e] = 0.0f;

  for (int k0 = 0; k0 < Ktot; k0 += 32) {
    // ---- stage B tile (im2col gather) into LDS hi/lo planes ----
    __syncthreads();
    {
      int kk = k0 + 2 * kp_base;
      int c  = (int)((unsigned)kk / 9u);
      int r9 = kk - 9 * c;
      #pragma unroll
      for (int i = 0; i < 4; ++i) {
        const int kp = kp_base + 4 * i;
        float v0 = 0.0f;
        {
          const int dh = (r9 * 11) >> 5, dw = r9 - 3 * dh;
          const int hh = h_ld + dh - 1, ww = w_ld + dw - 1;
          if (sv && (unsigned)hh < (unsigned)H && (unsigned)ww < (unsigned)Wsp)
            v0 = Xn[c * HW + pos_off + (dh - 1) * Wsp + (dw - 1)];
        }
        int r9b = r9 + 1, cb = c;
        if (r9b >= 9) { r9b -= 9; ++cb; }
        float v1 = 0.0f;
        {
          const int dh = (r9b * 11) >> 5, dw = r9b - 3 * dh;
          const int hh = h_ld + dh - 1, ww = w_ld + dw - 1;
          if (sv && (unsigned)hh < (unsigned)H && (unsigned)ww < (unsigned)Wsp)
            v1 = Xn[cb * HW + pos_off + (dh - 1) * Wsp + (dw - 1)];
        }
        unsigned h, l;
        split_pair(v0, v1, h, l);
        bhi_p[nn_ld * CSTR + kp] = h;
        blo_p[nn_ld * CSTR + kp] = l;
        r9 = r9b + 7; c = cb;
        if (r9 >= 9) { r9 -= 9; ++c; }
      }
    }
    __syncthreads();

    // ---- A fragments ----
    // dword v holds K-pair (k0/2 + 4*kgrp) + v + 4*(v>=4): two b128 runs
    v16bf ahi[MT], alo[MT];
    #pragma unroll
    for (int mt = 0; mt < MT; ++mt) {
      const int o = o_wave + mt * 16 + m_lane;
      Q2BF xh, xl;
      if (PRE) {
        if (o < O) {
          const unsigned* hp = Whi + (size_t)o * Kp + (k0 >> 1) + 4 * kgrp;
          const unsigned* lp = Wlo + (size_t)o * Kp + (k0 >> 1) + 4 * kgrp;
          xh.q[0] = *(const v4u*)hp; xh.q[1] = *(const v4u*)(hp + 8);
          xl.q[0] = *(const v4u*)lp; xl.q[1] = *(const v4u*)(lp + 8);
        } else {
          xh.q[0] = v4zero(); xh.q[1] = v4zero();
          xl.q[0] = v4zero(); xl.q[1] = v4zero();
        }
      } else {
        unsigned hu[8], lu[8];
        if (o < O) {
          const float* ap = Wt + (size_t)o * Ktot + k0 + kgrp * 8;
          #pragma unroll
          for (int v = 0; v < 8; ++v) {
            const int b = 2 * v + (v >= 4 ? 8 : 0);
            split_pair(ap[b], ap[b + 1], hu[v], lu[v]);
          }
        } else {
          #pragma unroll
          for (int v = 0; v < 8; ++v) { hu[v] = 0u; lu[v] = 0u; }
        }
        #pragma unroll
        for (int v = 0; v < 4; ++v) {
          xh.q[0][v] = hu[v]; xh.q[1][v] = hu[v + 4];
          xl.q[0][v] = lu[v]; xl.q[1][v] = lu[v + 4];
        }
      }
      ahi[mt] = xh.bf; alo[mt] = xl.bf;
    }

    // ---- B fragments: 2x ds_load_b128 per plane ----
    v16bf bhi[NT], blo[NT];
    #pragma unroll
    for (int nt = 0; nt < NT; ++nt) {
      const int nn   = nt * 16 + m_lane;
      const int base = nn * CSTR + 4 * kgrp;
      Q2BF xh, xl;
      xh.q[0] = *(const v4u*)&bhi_p[base];
      xh.q[1] = *(const v4u*)&bhi_p[base + 8];
      xl.q[0] = *(const v4u*)&blo_p[base];
      xl.q[1] = *(const v4u*)&blo_p[base + 8];
      bhi[nt] = xh.bf; blo[nt] = xl.bf;
    }

    // ---- 3-term split-bf16 WMMA ----
    #pragma unroll
    for (int mt = 0; mt < MT; ++mt)
      #pragma unroll
      for (int nt = 0; nt < NT; ++nt) {
        acc[mt][nt] = __builtin_amdgcn_wmma_f32_16x16x32_bf16(
            false, ahi[mt], false, bhi[nt], (short)0, acc[mt][nt], false, false);
        acc[mt][nt] = __builtin_amdgcn_wmma_f32_16x16x32_bf16(
            false, ahi[mt], false, blo[nt], (short)0, acc[mt][nt], false, false);
        acc[mt][nt] = __builtin_amdgcn_wmma_f32_16x16x32_bf16(
            false, alo[mt], false, bhi[nt], (short)0, acc[mt][nt], false, false);
      }
  }

  // ---- epilogue: bias + SSD permutation scatter ----
  #pragma unroll
  for (int nt = 0; nt < NT; ++nt) {
    const int s = s_base + nt * 16 + m_lane;
    if (s >= NHW) continue;
    const int n   = s / HW;
    const int pos = s - n * HW;
    #pragma unroll
    for (int mt = 0; mt < MT; ++mt) {
      #pragma unroll
      for (int rr = 0; rr < 8; ++rr) {
        const int o = o_wave + mt * 16 + rr + 8 * kgrp;
        if (o < O) {
          const int a   = o / num_cols;
          const int col = o - a * num_cols;
          const long long anchor = (long long)anchor_base + (long long)pos * A + a;
          const long long idx = out_base +
              ((long long)n * TOTAL_ANCHORS + anchor) * num_cols + col;
          Out[idx] = acc[mt][nt][rr] + Bias[o];
        }
      }
    }
  }
}

extern "C" void kernel_launch(void* const* d_in, const int* in_sizes, int n_in,
                              void* d_out, int out_size, void* d_ws, size_t ws_size,
                              hipStream_t stream) {
  static const int FS[6] = {38, 19, 10, 5, 3, 1};
  static const int CH[6] = {512, 1024, 512, 256, 256, 256};
  static const int NA[6] = {4, 6, 6, 6, 4, 4};
  static const int AB[6] = {0, 5776, 7942, 8542, 8692, 8728};

  const long long cls_base = (long long)BATCH * TOTAL_ANCHORS * 4;
  float* out = (float*)d_out;

  // total scratch dwords for packed weights (hi+lo = #elements)
  size_t need_dw = 0;
  for (int i = 0; i < 6; ++i) {
    const size_t Ktot = (size_t)CH[i] * 9;
    need_dw += (size_t)(4 * NA[i]) * Ktot + (size_t)(91 * NA[i]) * Ktot;
  }
  const bool pre = (ws_size >= need_dw * 4);
  unsigned* wsbase = (unsigned*)d_ws;
  size_t off = 0;

  for (int i = 0; i < 6; ++i) {
    const float* feat = (const float*)d_in[5 * i + 0];
    const float* cw   = (const float*)d_in[5 * i + 1];
    const float* cb   = (const float*)d_in[5 * i + 2];
    const float* rw   = (const float*)d_in[5 * i + 3];
    const float* rb   = (const float*)d_in[5 * i + 4];
    const int H = FS[i], C = CH[i], A = NA[i];
    const int Ktot = C * 9;
    const int NHW  = BATCH * H * H;
    const int gx   = (NHW + 31) / 32;

    for (int head = 0; head < 2; ++head) {
      const int   nc   = head ? 91 : 4;
      const int   O    = nc * A;
      const float* w   = head ? cw : rw;
      const float* b   = head ? cb : rb;
      const long long ob = head ? cls_base : 0LL;
      const int   gy   = (O + 127) / 128;

      if (pre) {
        const int npairs = O * (Ktot >> 1);
        unsigned* hi = wsbase + off;  off += (size_t)npairs;
        unsigned* lo = wsbase + off;  off += (size_t)npairs;
        pack_weights<<<(npairs + 255) / 256, 256, 0, stream>>>(w, hi, lo, npairs);
        ssd_conv3x3_wmma<true><<<dim3(gx, gy), 128, 0, stream>>>(
            feat, w, hi, lo, b, out, C, H, H, O, A, nc, AB[i], ob);
      } else {
        ssd_conv3x3_wmma<false><<<dim3(gx, gy), 128, 0, stream>>>(
            feat, w, nullptr, nullptr, b, out, C, H, H, O, A, nc, AB[i], ob);
      }
    }
  }
}
